// MultiHeadAttention_66322884984909
// MI455X (gfx1250) — matrix operand order repaired and checked by reading, not compiled, not run on hardware
//
#include <hip/hip_runtime.h>
#include <hip/hip_bf16.h>
#include <stdint.h>

typedef __bf16 bf16;
typedef __attribute__((ext_vector_type(16))) __bf16 v16bf;
typedef __attribute__((ext_vector_type(8)))  __bf16 v8bf;
typedef __attribute__((ext_vector_type(8)))  float  v8f;
typedef int v4i_ __attribute__((__vector_size__(4 * sizeof(int))));

#define B_   2
#define L_   2048
#define D_   1024
#define H_   16
#define HD_  64
#define M_   (B_ * L_)     // 4096 rows total

// ---------------------------------------------------------------------------
// CDNA5 async global->LDS staging (ASYNCcnt path), with safe fallback.
// Builtin signature (from hipcc diagnostic): (v4i AS1*, v4i AS3*, imm, imm).
// ---------------------------------------------------------------------------
#if defined(__HIP_DEVICE_COMPILE__) && __has_builtin(__builtin_amdgcn_global_load_async_to_lds_b128)
#define ASYNC_LDS 1
#else
#define ASYNC_LDS 0
#endif

typedef __attribute__((address_space(1))) v4i_* gptr_v4i;
typedef __attribute__((address_space(3))) v4i_* lptr_v4i;

__device__ __forceinline__ void stage_b128(const bf16* g, bf16* l) {
#if ASYNC_LDS
  // Flat LDS pointer low 32 bits == LDS offset (ISA 10.2 aperture rules).
  __builtin_amdgcn_global_load_async_to_lds_b128(
      (gptr_v4i)(uintptr_t)g,
      (lptr_v4i)(uint32_t)(uintptr_t)l,
      0, 0);
#else
  *(v8bf*)l = *(const v8bf*)g;
#endif
}

__device__ __forceinline__ void wait_async() {
#if ASYNC_LDS
#if __has_builtin(__builtin_amdgcn_s_wait_asynccnt)
  __builtin_amdgcn_s_wait_asynccnt(0);
#else
  asm volatile("s_wait_asynccnt 0" ::: "memory");
#endif
#endif
}

// ---------------------------------------------------------------------------
// Fragment helpers for V_WMMA_F32_16X16X32_BF16 (wave32).
//
// A fragment (16x32, MxK): lane m = lane&15, half hb = lane>>4.
//   VGPR v holds K pair (hb*16 + 2v, +1) => one contiguous 16-elem run at hb*16.
// B fragment (32x16, KxN): lane n = lane&15, half hb = lane>>4.
//   VGPR v holds K pair (hb*16 + 2v, +1) => one contiguous 16-elem run at hb*16.
// C/D (16x16 f32): VGPR r holds (M = r + 8*(lane>>4), N = lane&15).
// ---------------------------------------------------------------------------

__device__ __forceinline__ v16bf cat8(v8bf a, v8bf b) {
  v16bf r;
#pragma unroll
  for (int i = 0; i < 8; ++i) { r[i] = a[i]; r[8 + i] = b[i]; }
  return r;
}

// A fragment from row-major src (16 rows, 32 K-contiguous cols), ld in elements.
__device__ __forceinline__ v16bf load_a_frag(const bf16* src, int ld, int lane) {
  const int m  = lane & 15;
  const int hb = lane >> 4;
  const bf16* p = src + (size_t)m * ld + hb * 16;
  v8bf c0 = *(const v8bf*)(p);
  v8bf c1 = *(const v8bf*)(p + 8);
  return cat8(c0, c1);
}

// B fragment from a pointer at this lane's contiguous 16-element K-run
// (16-byte aligned is sufficient; used for padded LDS tiles).
__device__ __forceinline__ v16bf load_b16(const bf16* p) {
  v8bf c0 = *(const v8bf*)(p);
  v8bf c1 = *(const v8bf*)(p + 8);
  return cat8(c0, c1);
}

__device__ __forceinline__ v8f wmma_bf16(v16bf a, v16bf b, v8f c) {
  return __builtin_amdgcn_wmma_f32_16x16x32_bf16(false, a, false, b,
                                                 (short)0, c, false, false);
}

// ---------------------------------------------------------------------------
// f32 -> bf16 conversion
// ---------------------------------------------------------------------------
__global__ __launch_bounds__(256) void cvt_bf16_kernel(const float* __restrict__ in,
                                                       bf16* __restrict__ out, int n) {
  int i = blockIdx.x * 256 + threadIdx.x;
  if (i < n) out[i] = (bf16)in[i];
}

// ---------------------------------------------------------------------------
// NT GEMM: C[m,n] = sum_k A[m,k] * W[n,k]; M=4096, N=1024, K=1024.
// 256 threads = 8 waves; wave w owns rows [bx*128 + 16w, +16), cols [by*64, +64).
// Shared W tile (64 x 32 bf16) staged per k-step into double-buffered LDS via
// async-to-LDS; A fragments stream from global per wave (unique rows).
// MODE 0: store bf16 flat (Q,K).  MODE 1: store bf16 per-head-transposed (V).
// MODE 2: store f32 + bias (final output projection).
// ---------------------------------------------------------------------------
#define WSTR 40   // padded LDS row stride (bf16) for 64-row W/V tiles

template <int MODE>
__global__ __launch_bounds__(256) void gemm_nt_kernel(
    const bf16* __restrict__ A, const bf16* __restrict__ W,
    bf16* __restrict__ outb, float* __restrict__ outf,
    const float* __restrict__ bias) {
  __shared__ __align__(16) bf16 wbuf[2][64 * WSTR];  // 2 x 5120 B

  const int tid  = threadIdx.x;
  const int lane = tid & 31;
  const int w    = tid >> 5;
  const int m0   = blockIdx.x * 128 + w * 16;
  const int n0   = blockIdx.y * 64;

  const bf16* aRow = A + (size_t)m0 * D_;
  const bf16* wRow = W + (size_t)n0 * D_;

  // One 16B chunk per thread covers the whole 64x32 tile (256 chunks).
  const int srow = tid >> 2;           // 0..63
  const int scc  = (tid & 3) * 8;      // 0,8,16,24
  auto stageW = [&](int buf, int k0) {
    stage_b128(wRow + (size_t)srow * D_ + k0 + scc, &wbuf[buf][srow * WSTR + scc]);
  };

  stageW(0, 0);
  wait_async();
  __syncthreads();

  const int half = lane >> 4;
  const int nlo  = lane & 15;

  v8f acc[4] = {};
  for (int k0 = 0; k0 < D_; k0 += 32) {
    const int cur = (k0 >> 5) & 1;
    if (k0 + 32 < D_) stageW(cur ^ 1, k0 + 32);            // overlap DMA w/ math
    if (k0 + 64 < D_)
      __builtin_prefetch(aRow + (size_t)nlo * D_ + k0 + 64, 0, 1);

    v16bf a = load_a_frag(aRow + k0, D_, lane);
#pragma unroll
    for (int nb = 0; nb < 4; ++nb) {
      v16bf b = load_b16(&wbuf[cur][(nb * 16 + nlo) * WSTR + half * 16]);
      acc[nb] = wmma_bf16(a, b, acc[nb]);
    }
    wait_async();
    __syncthreads();
  }

#pragma unroll
  for (int nb = 0; nb < 4; ++nb) {
#pragma unroll
    for (int r = 0; r < 8; ++r) {
      const int m = m0 + r + 8 * half;
      const int n = n0 + nb * 16 + nlo;
      const float v = acc[nb][r];
      if (MODE == 0) {
        outb[(size_t)m * D_ + n] = (bf16)v;
      } else if (MODE == 1) {
        // Scatter V transposed per head: Vt[(b*H + h)*HD + j][l], l-major (len L).
        const int bb = m >> 11;           // / L
        const int lo = m & (L_ - 1);
        const int hh = lo >> 7;           // head = lo / 128 (quirky reshape)
        const int l  = (lo & 127) * 16 + (n >> 6);
        const int j  = n & (HD_ - 1);
        outb[((size_t)((bb * H_ + hh) * HD_ + j)) * L_ + l] = (bf16)v;
      } else {
        outf[(size_t)m * D_ + n] = v + bias[n];
      }
    }
  }
}

// ---------------------------------------------------------------------------
// Causal flash attention over one (b, h, 128-row block). 8 waves, 16 rows/wave.
// Q,K flat head-major bf16 (L x hd rows); V pre-transposed per head (hd x L).
// Block-uniform j-loop (bounded by the block causal limit) so all 8 waves share
// double-buffered K/V tiles staged into LDS with async-to-LDS. Fully masked
// tiles are no-ops in the online softmax (scale=1, p=0).
// ---------------------------------------------------------------------------
#define KSTR 72   // padded LDS row stride (bf16) for the 32x64 K tile

__global__ __launch_bounds__(256) void attn_kernel(
    const bf16* __restrict__ Q, const bf16* __restrict__ K,
    const bf16* __restrict__ Vt, bf16* __restrict__ C) {
  __shared__ __align__(16) bf16 kbuf[2][32 * KSTR];  // 2 x 4608 B
  __shared__ __align__(16) bf16 vbuf[2][64 * WSTR];  // 2 x 5120 B
  __shared__ __align__(16) bf16 pbuf[8][16 * WSTR];  // per-wave P tile, 10240 B

  const int tid  = threadIdx.x;
  const int lane = tid & 31;
  const int w    = tid >> 5;
  const int tile = blockIdx.x & 15;   // L/128 = 16 row-blocks per head
  const int bh   = blockIdx.x >> 4;
  const int h    = bh & (H_ - 1);
  const int b    = bh >> 4;
  const int i0   = tile * 128;
  const int m0   = i0 + w * 16;       // query row base within head

  const size_t headOff = (size_t)b * L_ * D_ + (size_t)h * L_ * HD_;
  const bf16* qh = Q + headOff;
  const bf16* kh = K + headOff;
  const bf16* vt = Vt + (size_t)(b * H_ + h) * HD_ * L_;
  bf16* ch = C + headOff;

  // Q strip (16 x 64) as two A fragments, kept in registers.
  v16bf qa[2];
#pragma unroll
  for (int kc = 0; kc < 2; ++kc)
    qa[kc] = load_a_frag(qh + (size_t)m0 * HD_ + kc * 32, HD_, lane);

  // Cooperative staging: 2 x 16B chunks per thread cover K(32x64) + Vt(64x32).
  const int kr  = tid >> 3;            // 0..31 (key row)
  const int kcc = (tid & 7) * 8;       // 0..56 (feature)
  const int vc  = tid >> 2;            // 0..63 (output col)
  const int vcc = (tid & 3) * 8;       // 0..24 (key)
  auto stageKV = [&](int buf, int j0) {
    stage_b128(kh + (size_t)(j0 + kr) * HD_ + kcc, &kbuf[buf][kr * KSTR + kcc]);
    stage_b128(vt + (size_t)vc * L_ + j0 + vcc,    &vbuf[buf][vc * WSTR + vcc]);
  };

  v8f o[4] = {};
  float mrow[8], lrow[8];
#pragma unroll
  for (int r = 0; r < 8; ++r) { mrow[r] = -__builtin_inff(); lrow[r] = 0.f; }

  const int half = lane >> 4;
  const int nlo  = lane & 15;
  const int jMax = i0 + 128;  // block-level causal bound (uniform across block)

  stageKV(0, 0);
  wait_async();
  __syncthreads();

  for (int j0 = 0; j0 < jMax; j0 += 32) {
    const int cur = (j0 >> 5) & 1;
    if (j0 + 32 < jMax) stageKV(cur ^ 1, j0 + 32);

    // S = Q K^T for 2 column tiles of 16 keys each, K from LDS.
    v8f s[2] = {};
#pragma unroll
    for (int c = 0; c < 2; ++c) {
#pragma unroll
      for (int kc = 0; kc < 2; ++kc) {
        v16bf kb = load_b16(&kbuf[cur][(16 * c + nlo) * KSTR + kc * 32 + half * 16]);
        s[c] = wmma_bf16(qa[kc], kb, s[c]);
      }
    }

    // Causal mask + online softmax; write P (bf16) to per-wave LDS tile.
#pragma unroll
    for (int r = 0; r < 8; ++r) {
      const int row = m0 + r + 8 * half;
      float s0 = ((j0 + nlo) > row)      ? -__builtin_inff() : s[0][r];
      float s1 = ((j0 + 16 + nlo) > row) ? -__builtin_inff() : s[1][r];
      float rmax = fmaxf(s0, s1);
#pragma unroll
      for (int off = 1; off < 16; off <<= 1)
        rmax = fmaxf(rmax, __shfl_xor(rmax, off, 16));
      const float newM  = fmaxf(mrow[r], rmax);
      const float scale = __expf(mrow[r] - newM);
      const float p0 = __expf(s0 - newM);
      const float p1 = __expf(s1 - newM);
      float psum = p0 + p1;
#pragma unroll
      for (int off = 1; off < 16; off <<= 1)
        psum += __shfl_xor(psum, off, 16);
      lrow[r] = lrow[r] * scale + psum;
      mrow[r] = newM;
#pragma unroll
      for (int nb = 0; nb < 4; ++nb) o[nb][r] *= scale;
      const int lr = r + 8 * half;  // local row 0..15
      pbuf[w][lr * WSTR + nlo]      = (bf16)p0;
      pbuf[w][lr * WSTR + 16 + nlo] = (bf16)p1;
    }
    asm volatile("s_wait_dscnt 0" ::: "memory");  // in-wave LDS RAW on P

    // Re-load P in A-fragment layout; O += P (16x32) * V (32x64), V from LDS.
    {
      const bf16* pp = &pbuf[w][nlo * WSTR + half * 16];
      v8bf c0 = *(const v8bf*)(pp);
      v8bf c1 = *(const v8bf*)(pp + 8);
      v16bf pa = cat8(c0, c1);
#pragma unroll
      for (int nb = 0; nb < 4; ++nb) {
        v16bf vb = load_b16(&vbuf[cur][(nb * 16 + nlo) * WSTR + half * 16]);
        o[nb] = wmma_bf16(pa, vb, o[nb]);
      }
    }

    wait_async();     // own-wave async chunks for the next buffer are done
    __syncthreads();  // ... and visible to all waves before anyone reads them
  }

  // Normalize and store context (bf16, flat head-major == (B,L,D) flat view).
#pragma unroll
  for (int r = 0; r < 8; ++r) {
    const float inv = 1.f / lrow[r];
    const int l = m0 + r + 8 * half;
#pragma unroll
    for (int nb = 0; nb < 4; ++nb)
      ch[(size_t)l * HD_ + nb * 16 + nlo] = (bf16)(o[nb][r] * inv);
  }
}

// ---------------------------------------------------------------------------
// Host-side orchestration.
// ---------------------------------------------------------------------------
extern "C" void kernel_launch(void* const* d_in, const int* in_sizes, int n_in,
                              void* d_out, int out_size, void* d_ws, size_t ws_size,
                              hipStream_t stream) {
  (void)in_sizes; (void)n_in; (void)out_size; (void)ws_size;

  const float* X  = (const float*)d_in[0];  // (B,L,D)
  const float* Wq = (const float*)d_in[1];
  const float* Wk = (const float*)d_in[2];
  const float* Wv = (const float*)d_in[3];
  const float* Wo = (const float*)d_in[4];
  const float* bO = (const float*)d_in[5];
  float* Y = (float*)d_out;

  const size_t NX = (size_t)M_ * D_;   // 4M elements
  const size_t NW = (size_t)D_ * D_;   // 1M elements

  bf16* Xb  = (bf16*)d_ws;
  bf16* Wqb = Xb + NX;
  bf16* Wkb = Wqb + NW;
  bf16* Wvb = Wkb + NW;
  bf16* Wob = Wvb + NW;
  bf16* Qb  = Wob + NW;
  bf16* Kb  = Qb + NX;
  bf16* Vt  = Kb + NX;
  bf16* Cb  = Vt + NX;   // total 24M bf16 = 48 MB of workspace

  cvt_bf16_kernel<<<(int)((NX + 255) / 256), 256, 0, stream>>>(X, Xb, (int)NX);
  cvt_bf16_kernel<<<(int)((NW + 255) / 256), 256, 0, stream>>>(Wq, Wqb, (int)NW);
  cvt_bf16_kernel<<<(int)((NW + 255) / 256), 256, 0, stream>>>(Wk, Wkb, (int)NW);
  cvt_bf16_kernel<<<(int)((NW + 255) / 256), 256, 0, stream>>>(Wv, Wvb, (int)NW);
  cvt_bf16_kernel<<<(int)((NW + 255) / 256), 256, 0, stream>>>(Wo, Wob, (int)NW);

  dim3 gg(M_ / 128, D_ / 64);  // 32 x 16 workgroups
  gemm_nt_kernel<0><<<gg, 256, 0, stream>>>(Xb, Wqb, Qb, nullptr, nullptr);
  gemm_nt_kernel<0><<<gg, 256, 0, stream>>>(Xb, Wkb, Kb, nullptr, nullptr);
  gemm_nt_kernel<1><<<gg, 256, 0, stream>>>(Xb, Wvb, Vt, nullptr, nullptr);

  attn_kernel<<<B_ * H_ * (L_ / 128), 256, 0, stream>>>(Qb, Kb, Vt, Cb);

  gemm_nt_kernel<2><<<gg, 256, 0, stream>>>(Cb, Wob, nullptr, Y, bO);
}
